// GraphNet_77687368450202
// MI455X (gfx1250) — matrix operand order repaired and checked by reading, not compiled, not run on hardware
//
#include <hip/hip_runtime.h>
#include <hip/hip_bf16.h>

// ---------------- problem constants ----------------
#define NNODES   297
#define BATCH    128
#define WALK     8
#define NPROWS   (BATCH * NNODES)   // 38016, divisible by 16
#define LDK      320                // padded feature stride (19*16=304 padded to 320 for K loop)
#define NPADR    304                // padded row count (19*16)
#define NTILES   19                 // 304/16 N tiles
#define MTILES_P (NPROWS / 16)      // 2376
#define MTILES_N (NPADR / 16)       // 19
#define KSLAB    32                 // K elements staged in LDS per step
#define SLAB_BYTES (NPADR * KSLAB * 2)          // 19456 B
#define SLAB_U4    (SLAB_BYTES / 16)            // 1216 uint4 chunks

typedef __attribute__((ext_vector_type(16))) __bf16 v16bf;
typedef __attribute__((ext_vector_type(8)))  __bf16 v8bf;
typedef __attribute__((ext_vector_type(4)))  __bf16 v4bf;
typedef __attribute__((ext_vector_type(8)))  float  v8f;

static __device__ inline __bf16 f2bf(float f) {
    union { float f; unsigned u; } v; v.f = f;
    unsigned r = v.u + 0x7FFFu + ((v.u >> 16) & 1u);   // round-to-nearest-even
    unsigned short h = (unsigned short)(r >> 16);
    return __builtin_bit_cast(__bf16, h);
}

// ---------------- utility kernels ----------------
__global__ void zero4_kernel(float4* p, size_t n4) {
    size_t stride = (size_t)gridDim.x * blockDim.x;
    for (size_t i = (size_t)blockIdx.x * blockDim.x + threadIdx.x; i < n4; i += stride)
        p[i] = make_float4(0.f, 0.f, 0.f, 0.f);
}

__global__ void pad_copy_kernel(const float* __restrict__ src, float* __restrict__ dst, int rows) {
    int total = rows * NNODES;
    int stride = gridDim.x * blockDim.x;
    for (int i = blockIdx.x * blockDim.x + threadIdx.x; i < total; i += stride) {
        int r = i / NNODES, c = i - r * NNODES;
        dst[(size_t)r * LDK + c] = src[i];
    }
}

__global__ void convert_w_kernel(const float* __restrict__ W, __bf16* __restrict__ Wb) {
    int idx = blockIdx.x * blockDim.x + threadIdx.x;   // over NPADR*LDK
    if (idx >= NPADR * LDK) return;
    int j = idx / LDK, k = idx - j * LDK;
    float v = (j < NNODES && k < NNODES) ? W[j * NNODES + k] : 0.0f;
    Wb[idx] = f2bf(v);
}

// f32 H -> bf16 Hb (same padded row-major layout), 4 elements per thread-iter
__global__ void h2b_kernel(const float* __restrict__ H, __bf16* __restrict__ Hb, size_t n) {
    size_t stride = (size_t)gridDim.x * blockDim.x * 4;
    for (size_t i = ((size_t)blockIdx.x * blockDim.x + threadIdx.x) * 4; i < n; i += stride) {
        float4 v = *(const float4*)(H + i);
        v4bf o;
        o[0] = f2bf(v.x); o[1] = f2bf(v.y); o[2] = f2bf(v.z); o[3] = f2bf(v.w);
        *(v4bf*)(Hb + i) = o;
    }
}

__global__ void consts_kernel(float* s2, float* inv, float* one) {
    *s2 = 0.0f; *inv = 1.0f; *one = 1.0f;
}

// ---------------- WMMA GEMM: Y = H @ W^T  (bf16 in, f32 accumulate) ----------------
// block = 256 (8 waves); each wave owns one 16-row M strip and all 19 N tiles.
// Per K-step: the block stages W[:, k0:k0+32] into LDS, each wave loads one A
// fragment and issues 19 back-to-back v_wmma_f32_16x16x32_bf16.
__global__ __launch_bounds__(256)
void wmma_gemm_kernel(const __bf16* __restrict__ Hb, const __bf16* __restrict__ Wb,
                      float* __restrict__ Y, int Mtiles) {
    extern __shared__ char smem[];
    __bf16* sW = (__bf16*)smem;                 // [NPADR][KSLAB] bf16 slab

    const int wave = threadIdx.x >> 5;
    const int tm   = blockIdx.x * 8 + wave;
    const bool active = (tm < Mtiles);
    const int lane = threadIdx.x & 31;
    const int half = lane >> 4;                 // 0: lanes 0-15, 1: lanes 16-31
    const int l    = lane & 15;

    // A fragment base: lane(half=0) holds K {0..15}, lane(half=1) K {16..31}
    const __bf16* __restrict__ Ar =
        Hb + (size_t)((active ? tm : 0) * 16 + l) * LDK + half * 16;

    v8f acc[NTILES];
#pragma unroll
    for (int t = 0; t < NTILES; ++t) {
        v8f z = {0.f, 0.f, 0.f, 0.f, 0.f, 0.f, 0.f, 0.f};
        acc[t] = z;
    }

    for (int k0 = 0; k0 < LDK; k0 += KSLAB) {
        // --- cooperative stage of W[:, k0:k0+32] into LDS (row stride KSLAB) ---
        {
            uint4* dst = (uint4*)sW;
            for (int c = threadIdx.x; c < SLAB_U4; c += 256) {
                int r = c >> 2, part = c & 3;   // 4 uint4 per row of 32 bf16
                dst[c] = *((const uint4*)(Wb + (size_t)r * LDK + k0) + part);
            }
        }
        __syncthreads();

        if (active) {
            __builtin_prefetch((const void*)(Ar + k0 + KSLAB), 0, 0);  // global_prefetch_b8
            v8bf a_lo = *(const v8bf*)(Ar + k0);
            v8bf a_hi = *(const v8bf*)(Ar + k0 + 8);
            v16bf A = __builtin_shufflevector(a_lo, a_hi,
                        0,1,2,3,4,5,6,7,8,9,10,11,12,13,14,15);
#pragma unroll
            for (int tn = 0; tn < NTILES; ++tn) {
                // B[k][n] = W[n][k]; lane n reads 16 consecutive bf16 of row n from LDS
                const v16bf B = *(const v16bf*)(sW + ((tn * 16 + l) * KSLAB + half * 16));
                acc[tn] = __builtin_amdgcn_wmma_f32_16x16x32_bf16(
                              false, A, false, B, (short)0, acc[tn], false, false);
            }
        }
        __syncthreads();
    }

    if (active) {
        // D layout: VGPR r -> row (r + 8*half), col = lane&15
        float* Yb = Y + (size_t)(tm * 16 + half * 8) * LDK + l;
#pragma unroll
        for (int tn = 0; tn < NTILES; ++tn)
#pragma unroll
            for (int r = 0; r < 8; ++r)
                Yb[(size_t)r * LDK + tn * 16] = acc[tn][r];
    }
}

// ---------------- edge scatter: Hout[dst] += Y[src] * inv_scale ----------------
__global__ void scatter_kernel(const int* __restrict__ esrc, const int* __restrict__ edst,
                               const float* __restrict__ Y, float* __restrict__ Hout,
                               int nE, const float* __restrict__ invp) {
    const float inv = *invp;
    const int wave = (blockIdx.x * blockDim.x + threadIdx.x) >> 5;
    const int lane = threadIdx.x & 31;
    if (wave >= nE) return;
    const size_t srow = (size_t)esrc[wave] * LDK;
    const size_t drow = (size_t)edst[wave] * LDK;
    for (int j = lane; j < NNODES; j += 32)
        unsafeAtomicAdd(&Hout[drow + j], Y[srow + j] * inv);
}

// ---------------- scale = ||h_p[:,0]||, inv = 1/scale ----------------
__global__ void col0_sumsq_kernel(const float* __restrict__ Hp, float* accum) {
    float v = 0.0f;
    int stride = gridDim.x * blockDim.x;
    for (int m = blockIdx.x * blockDim.x + threadIdx.x; m < NPROWS; m += stride) {
        float x = Hp[(size_t)m * LDK];
        v += x * x;
    }
    for (int off = 16; off > 0; off >>= 1) v += __shfl_down(v, off, 32);
    if ((threadIdx.x & 31) == 0) unsafeAtomicAdd(accum, v);
}

__global__ void finalize_scale_kernel(const float* s2, float* inv) { *inv = rsqrtf(*s2); }

__global__ void scale_inplace_kernel(float* p, size_t n, const float* __restrict__ invp) {
    const float inv = *invp;
    size_t stride = (size_t)gridDim.x * blockDim.x;
    for (size_t i = (size_t)blockIdx.x * blockDim.x + threadIdx.x; i < n; i += stride)
        p[i] *= inv;
}

// ---------------- trace readouts ----------------
__global__ void trace_big_kernel(const float* __restrict__ Hp, float* __restrict__ out) {
    const int b = blockIdx.x;               // 0..127
    __shared__ float sd[128];
    float v = 0.0f;
    for (int i = threadIdx.x; i < NNODES; i += 128)
        v += Hp[(size_t)(b * NNODES + i) * LDK + i];
    sd[threadIdx.x] = v; __syncthreads();
    for (int off = 64; off > 0; off >>= 1) {
        if ((int)threadIdx.x < off) sd[threadIdx.x] += sd[threadIdx.x + off];
        __syncthreads();
    }
    if (threadIdx.x == 0) out[b] = sd[0];
}

__global__ void trace_small_kernel(const float* __restrict__ Hn, float* __restrict__ out) {
    __shared__ float sd[128];
    float v = 0.0f;
    for (int i = threadIdx.x; i < NNODES; i += 128)
        v += Hn[(size_t)i * LDK + i];
    sd[threadIdx.x] = v; __syncthreads();
    for (int off = 64; off > 0; off >>= 1) {
        if ((int)threadIdx.x < off) sd[threadIdx.x] += sd[threadIdx.x + off];
        __syncthreads();
    }
    if (threadIdx.x == 0) *out = sd[0];
}

// ---------------- final: p_list, mean/std(ddof=1), z-score ----------------
__global__ void final_kernel(const float* __restrict__ trp, const float* __restrict__ trn,
                             const float* __restrict__ y, float* __restrict__ out) {
    __shared__ float s1[128], s2[128];
    const int b = threadIdx.x;              // 0..127
    const float sign = (y[b] - 0.5f) * 2.0f;
    float p[WALK];
#pragma unroll
    for (int t = 0; t < WALK; ++t)
        p[t] = (trp[t * BATCH + b] - trn[t]) * 100.0f * sign;
    for (int t = 0; t < WALK; ++t) {
        s1[b] = p[t]; s2[b] = p[t] * p[t]; __syncthreads();
        for (int off = 64; off > 0; off >>= 1) {
            if (b < off) { s1[b] += s1[b + off]; s2[b] += s2[b + off]; }
            __syncthreads();
        }
        float sum = s1[0], sumsq = s2[0];
        float mu  = sum / (float)BATCH;
        float var = (sumsq - sum * sum / (float)BATCH) / (float)(BATCH - 1);
        float sd  = sqrtf(var);
        __syncthreads();
        out[b * WALK + t] = (p[t] - mu) / sd;
    }
}

// ---------------- host-side orchestration ----------------
extern "C" void kernel_launch(void* const* d_in, const int* in_sizes, int n_in,
                              void* d_out, int out_size, void* d_ws, size_t ws_size,
                              hipStream_t stream) {
    const float* x_p  = (const float*)d_in[0];
    const float* x_np = (const float*)d_in[1];
    const float* y    = (const float*)d_in[2];
    const float* W    = (const float*)d_in[3];
    const int*   eip  = (const int*)d_in[4];
    const int*   ein  = (const int*)d_in[5];
    const int Ep = in_sizes[4] / 2;
    const int En = in_sizes[5] / 2;
    const int* sp = eip;       const int* dp = eip + Ep;
    const int* sn = ein;       const int* dn = ein + En;

    // workspace carve-up (chunks up to trn start 256B-aligned; s2/inv/one do not)
    const size_t fHp = (size_t)NPROWS * LDK;     // 12,165,120 floats
    const size_t fHn = (size_t)NPADR  * LDK;     //     97,280 floats
    char* pws = (char*)d_ws;
    float*  Hp  = (float*)pws;  pws += fHp * 4;
    float*  Yp  = (float*)pws;  pws += fHp * 4;
    float*  Hn  = (float*)pws;  pws += fHn * 4;
    float*  Yn  = (float*)pws;  pws += fHn * 4;
    __bf16* Hbp = (__bf16*)pws; pws += fHp * 2;
    __bf16* Hbn = (__bf16*)pws; pws += fHn * 2;
    __bf16* Wb  = (__bf16*)pws; pws += (size_t)NPADR * LDK * 2;
    float*  trp = (float*)pws;  pws += WALK * BATCH * 4;
    float*  trn = (float*)pws;  pws += WALK * 4;
    float*  s2  = (float*)pws;  pws += 4;
    float*  inv = (float*)pws;  pws += 4;
    float*  one = (float*)pws;  pws += 4;

    const int TPB = 256;
    const int zb_p = 4096;
    const int gemm_blk_p = (MTILES_P + 7) / 8;   // 297
    const int gemm_blk_n = (MTILES_N + 7) / 8;   // 3
    const int scat_blk_p = (Ep * 32 + TPB - 1) / TPB;
    const int scat_blk_n = (En * 32 + TPB - 1) / TPB;

    // one conv round: H = scatter(H @ W^T) * (*invp)
    auto conv_p = [&](const float* invp) {
        h2b_kernel<<<2048, TPB, 0, stream>>>(Hp, Hbp, fHp);
        wmma_gemm_kernel<<<gemm_blk_p, TPB, SLAB_BYTES, stream>>>(Hbp, Wb, Yp, MTILES_P);
        zero4_kernel<<<zb_p, TPB, 0, stream>>>((float4*)Hp, fHp / 4);
        scatter_kernel<<<scat_blk_p, TPB, 0, stream>>>(sp, dp, Yp, Hp, Ep, invp);
    };
    auto conv_n = [&](const float* invp) {
        h2b_kernel<<<32, TPB, 0, stream>>>(Hn, Hbn, fHn);
        wmma_gemm_kernel<<<gemm_blk_n, TPB, SLAB_BYTES, stream>>>(Hbn, Wb, Yn, MTILES_N);
        zero4_kernel<<<64, TPB, 0, stream>>>((float4*)Hn, fHn / 4);
        scatter_kernel<<<scat_blk_n, TPB, 0, stream>>>(sn, dn, Yn, Hn, En, invp);
    };

    // ---- setup ----
    consts_kernel<<<1, 1, 0, stream>>>(s2, inv, one);
    convert_w_kernel<<<(NPADR * LDK + TPB - 1) / TPB, TPB, 0, stream>>>(W, Wb);
    zero4_kernel<<<zb_p, TPB, 0, stream>>>((float4*)Hp, fHp / 4);
    pad_copy_kernel<<<8192, TPB, 0, stream>>>(x_p, Hp, NPROWS);
    zero4_kernel<<<64, TPB, 0, stream>>>((float4*)Hn, fHn / 4);
    pad_copy_kernel<<<(NNODES * NNODES + TPB - 1) / TPB, TPB, 0, stream>>>(x_np, Hn, NNODES);

    // ---- conv 1 (unscaled), then scale factor from column 0 of h_p ----
    conv_p(one);
    conv_n(one);
    col0_sumsq_kernel<<<(NPROWS + TPB - 1) / TPB, TPB, 0, stream>>>(Hp, s2);
    finalize_scale_kernel<<<1, 1, 0, stream>>>(s2, inv);
    scale_inplace_kernel<<<zb_p, TPB, 0, stream>>>(Hp, fHp, inv);
    scale_inplace_kernel<<<64, TPB, 0, stream>>>(Hn, fHn, inv);

    // ---- conv 2 (scaled) ----
    conv_p(inv);
    conv_n(inv);

    // ---- 8 walk steps with trace readouts ----
    for (int t = 0; t < WALK; ++t) {
        conv_p(inv);
        trace_big_kernel<<<BATCH, 128, 0, stream>>>(Hp, trp + t * BATCH);
        conv_n(inv);
        trace_small_kernel<<<1, 128, 0, stream>>>(Hn, trn + t);
    }

    // ---- z-scored readout -> d_out [128, 8] ----
    final_kernel<<<1, 128, 0, stream>>>(trp, trn, y, (float*)d_out);
}
